// SpatAttn_6107443495642
// MI455X (gfx1250) — compile-verified, matrix operand order checked by reading, not run on hardware
//
#include <hip/hip_runtime.h>

// Problem constants (from reference)
#define Bb   4
#define Cc   64
#define CQ   8
#define Nn   4096          // H*W = 64*64
#define QT   128           // queries per block (8 waves * 16)
#define WQ   16            // queries per wave
#define KC   128           // keys per LDS V chunk
#define NKC  (Nn / KC)     // 32 chunks

typedef __attribute__((ext_vector_type(16))) __bf16 v16bf;
typedef __attribute__((ext_vector_type(8)))  float  v8f;

typedef __attribute__((address_space(3))) unsigned int lds_u32;

__device__ __forceinline__ unsigned int lds_byte_offset(const void* p) {
    // flat -> local addrspace cast yields the wave-relative LDS byte address
    return (unsigned int)(unsigned long long)(lds_u32*)p;
}

__device__ __forceinline__ unsigned int f2bf_pair(float lo, float hi) {
    unsigned int a = __float_as_uint(lo);
    unsigned int b = __float_as_uint(hi);
    a = (a + 0x7FFFu + ((a >> 16) & 1u)) >> 16;            // RNE to bf16, low half
    b = (b + 0x7FFFu + ((b >> 16) & 1u)) & 0xFFFF0000u;    // RNE to bf16, high half
    return b | a;
}

__device__ __forceinline__ unsigned short f2bf(float x) {
    unsigned int a = __float_as_uint(x);
    return (unsigned short)((a + 0x7FFFu + ((a >> 16) & 1u)) >> 16);
}

// ---------------------------------------------------------------------------
// Kernel 1: 1x1-conv projections. One thread per (b, pixel n).
//   Qg, Kt : [B][N][8]  bf16  (row layout == WMMA operand dword layout)
//   Vg     : [B][C][N]  bf16
// ---------------------------------------------------------------------------
__global__ __launch_bounds__(256) void SpatAttn_proj_kernel(
    const float* __restrict__ x,
    const float* __restrict__ Wq, const float* __restrict__ bq,
    const float* __restrict__ Wk, const float* __restrict__ bk,
    const float* __restrict__ Wv, const float* __restrict__ bv,
    unsigned short* __restrict__ Qg,
    unsigned short* __restrict__ Kt,
    unsigned short* __restrict__ Vg)
{
    const int t = blockIdx.x * blockDim.x + threadIdx.x;   // 0 .. B*N-1
    const int b = t >> 12;
    const int n = t & (Nn - 1);
    const float* xb = x + (size_t)b * Cc * Nn + n;

    float q[CQ], k[CQ], v[Cc];
#pragma unroll
    for (int o = 0; o < CQ; ++o) { q[o] = bq[o]; k[o] = bk[o]; }
#pragma unroll
    for (int o = 0; o < Cc; ++o) { v[o] = bv[o]; }

#pragma unroll 4
    for (int c = 0; c < Cc; ++c) {
        float xc = xb[(size_t)c * Nn];                     // coalesced over n
#pragma unroll
        for (int o = 0; o < CQ; ++o) {
            q[o] = fmaf(Wq[o * Cc + c], xc, q[o]);         // scalar (uniform) weight loads
            k[o] = fmaf(Wk[o * Cc + c], xc, k[o]);
        }
#pragma unroll
        for (int o = 0; o < Cc; ++o)
            v[o] = fmaf(Wv[o * Cc + c], xc, v[o]);
    }

    unsigned short* qp = Qg + (size_t)t * CQ;
    unsigned short* kp = Kt + (size_t)t * CQ;
#pragma unroll
    for (int o = 0; o < CQ; ++o) { qp[o] = f2bf(q[o]); kp[o] = f2bf(k[o]); }

    unsigned short* vp = Vg + (size_t)b * Cc * Nn + n;
#pragma unroll
    for (int o = 0; o < Cc; ++o) vp[(size_t)o * Nn] = f2bf(v[o]);
}

// ---------------------------------------------------------------------------
// Async staging of one 64ch x 128key V chunk (bf16 pairs) into an LDS buffer.
// 1024 b128 transfers; each of 256 threads issues 4. Tracked by ASYNCcnt.
// ---------------------------------------------------------------------------
__device__ __forceinline__ void async_load_vchunk(const unsigned int* __restrict__ src,
                                                  unsigned int* dstbuf, int tid)
{
#pragma unroll
    for (int i = 0; i < 4; ++i) {
        int idx4 = tid + i * 256;                 // which 4-dword unit (1024 total)
        int c    = idx4 >> 4;                     // 16 units per 64-dword channel row
        int p4   = (idx4 & 15) << 2;              // dword offset within row
        const unsigned int* g = src + (size_t)c * (Nn / 2) + p4;
        unsigned int lds = lds_byte_offset(dstbuf + (idx4 << 2));
        asm volatile("global_load_async_to_lds_b128 %0, %1, off"
                     :: "v"(lds), "v"((unsigned long long)(const void*)g)
                     : "memory");
    }
}

__device__ __forceinline__ void wait_async_done() {
    asm volatile("s_wait_asynccnt 0x0" ::: "memory");
}

// ---------------------------------------------------------------------------
// Kernel 2: fused flash attention.
//   Sᵀ tile = Kᵀ(keys x 8,pad32) x Qᵀ(8,pad32 x queries)  -> C/D layout whose
//   per-lane contents pack directly into the A operand of the P·Vᵀ bf16 WMMA.
//   V chunks double-buffered in LDS via async-to-LDS loads (ASYNCcnt).
// ---------------------------------------------------------------------------
__global__ __launch_bounds__(256) void SpatAttn_flash_kernel(
    const unsigned int* __restrict__ Qg32,   // [B][N][4] dwords (bf16 pairs)
    const unsigned int* __restrict__ Kt32,   // [B][N][4]
    const unsigned int* __restrict__ Vg32,   // [B][C][N/2] dwords
    const float* __restrict__ gamma,
    float* __restrict__ out)                 // [B][C][N] f32
{
    __shared__ __align__(16) unsigned int Vl[2][Cc * (KC / 2)];  // 2 x 16 KB

    const int tid  = threadIdx.x;
    const int lane = tid & 31;
    const int wid  = tid >> 5;
    const int half = lane >> 4;              // 0 / 1
    const int l15  = lane & 15;
    const int b    = blockIdx.x >> 5;        // 32 query-blocks per batch
    const int qbase = ((blockIdx.x & 31) * QT) + wid * WQ;

    // Fixed B operand for energy: Qᵀ (cq padded to 32). lanes>=16 supply zeros.
    unsigned int bqv[8];
#pragma unroll
    for (int j = 0; j < 8; ++j) bqv[j] = 0u;
    if (half == 0) {
        const unsigned int* qp = Qg32 + ((size_t)(b * Nn + qbase + l15) << 2);
#pragma unroll
        for (int j = 0; j < 4; ++j) bqv[j] = qp[j];
    }

    v8f O[4];
#pragma unroll
    for (int ct = 0; ct < 4; ++ct)
#pragma unroll
        for (int r = 0; r < 8; ++r) O[ct][r] = 0.0f;

    float mrow = -3.0e38f;   // running max for this lane's query
    float lrow = 0.0f;       // running sum

    const unsigned int* Vbase = Vg32 + (size_t)b * Cc * (Nn / 2);

    // Preload chunk 0 asynchronously.
    async_load_vchunk(Vbase, &Vl[0][0], tid);

    for (int kc = 0; kc < NKC; ++kc) {
        wait_async_done();        // this wave's async writes to LDS landed
        __syncthreads();          // everyone's landed; prior buffer fully read

        if (kc + 1 < NKC)         // stream next chunk while computing this one
            async_load_vchunk(Vbase + (kc + 1) * (KC / 2),
                              &Vl[(kc + 1) & 1][0], tid);

        const unsigned int* Vcur = &Vl[kc & 1][0];

#pragma unroll 1
        for (int kb = 0; kb < KC / 32; ++kb) {
            // ---- energy: two 16-key Sᵀ tiles via bf16 WMMA (cq zero-padded) ----
            v8f s[2];
#pragma unroll
            for (int h = 0; h < 2; ++h) {
                union { unsigned int u[8]; v16bf v; } A, Bq;
#pragma unroll
                for (int j = 0; j < 8; ++j) { A.u[j] = 0u; Bq.u[j] = bqv[j]; }
                if (half == 0) {
                    int kg = kc * KC + kb * 32 + h * 16 + l15;
                    const unsigned int* kp = Kt32 + ((size_t)(b * Nn + kg) << 2);
#pragma unroll
                    for (int j = 0; j < 4; ++j) A.u[j] = kp[j];
                }
                v8f z;
#pragma unroll
                for (int r = 0; r < 8; ++r) z[r] = 0.0f;
                s[h] = __builtin_amdgcn_wmma_f32_16x16x32_bf16(
                           false, A.v, false, Bq.v, (short)0, z, false, false);
            }

            // ---- online softmax over the 32 keys (query = lane&15) ----
            float bm = s[0][0];
#pragma unroll
            for (int r = 1; r < 8; ++r) bm = fmaxf(bm, s[0][r]);
#pragma unroll
            for (int r = 0; r < 8; ++r) bm = fmaxf(bm, s[1][r]);
            bm = fmaxf(bm, __shfl_xor(bm, 16, 32));
            float mnew  = fmaxf(mrow, bm);
            float scale = __expf(mrow - mnew);
            float rs = 0.0f;
            float e0[8], e1[8];
#pragma unroll
            for (int r = 0; r < 8; ++r) { e0[r] = __expf(s[0][r] - mnew); rs += e0[r]; }
#pragma unroll
            for (int r = 0; r < 8; ++r) { e1[r] = __expf(s[1][r] - mnew); rs += e1[r]; }
            rs += __shfl_xor(rs, 16, 32);
            lrow = lrow * scale + rs;
            mrow = mnew;

            // ---- rescale O by per-query factor (broadcast from stat lanes) ----
#pragma unroll
            for (int r = 0; r < 8; ++r) {
                float f = __shfl(scale, r + 8 * half, 32);
                O[0][r] *= f; O[1][r] *= f; O[2][r] *= f; O[3][r] *= f;
            }

            // ---- pack P into A operand (Sᵀ layout -> pairwise bf16 pack) ----
            union { unsigned int u[8]; v16bf v; } P;
#pragma unroll
            for (int r = 0; r < 4; ++r) P.u[r]     = f2bf_pair(e0[2 * r], e0[2 * r + 1]);
#pragma unroll
            for (int r = 0; r < 4; ++r) P.u[4 + r] = f2bf_pair(e1[2 * r], e1[2 * r + 1]);

            // ---- O += P (16q x 32k) * Vᵀ (32k x 16c), 4 channel tiles ----
#pragma unroll
            for (int ct = 0; ct < 4; ++ct) {
                union { unsigned int u[8]; v16bf v; } Bv;
                int c = ct * 16 + l15;
#pragma unroll
                for (int r = 0; r < 8; ++r) {
                    int pair = kb * 16 + ((r >> 2) << 3) + (half << 2) + (r & 3);
                    Bv.u[r] = Vcur[c * (KC / 2) + pair];
                }
                O[ct] = __builtin_amdgcn_wmma_f32_16x16x32_bf16(
                            false, P.v, false, Bv.v, (short)0, O[ct], false, false);
            }
        }
        __syncthreads();          // all waves done with Vcur before it is rewritten
    }

    // ---- finalize: out[b][c][m] = gamma * O[m][c] / l[m] ----
    const float g = gamma[0];
    const float rfac = g / lrow;
    float* ob = out + (size_t)b * Cc * Nn;
#pragma unroll
    for (int r = 0; r < 8; ++r) {
        float f = __shfl(rfac, r + 8 * half, 32);
        int   m = qbase + r + 8 * half;
        ob[(size_t)(l15)      * Nn + m] = O[0][r] * f;
        ob[(size_t)(16 + l15) * Nn + m] = O[1][r] * f;
        ob[(size_t)(32 + l15) * Nn + m] = O[2][r] * f;
        ob[(size_t)(48 + l15) * Nn + m] = O[3][r] * f;
    }
}

// ---------------------------------------------------------------------------
extern "C" void kernel_launch(void* const* d_in, const int* in_sizes, int n_in,
                              void* d_out, int out_size, void* d_ws, size_t ws_size,
                              hipStream_t stream) {
    const float* x     = (const float*)d_in[0];
    const float* Wq    = (const float*)d_in[1];
    const float* bq    = (const float*)d_in[2];
    const float* Wk    = (const float*)d_in[3];
    const float* bk    = (const float*)d_in[4];
    const float* Wv    = (const float*)d_in[5];
    const float* bv    = (const float*)d_in[6];
    const float* gamma = (const float*)d_in[7];
    float* out = (float*)d_out;

    // Workspace: Q (256KB) + Kt (256KB) + V (2MB), all bf16
    unsigned short* Qg = (unsigned short*)d_ws;
    unsigned short* Kt = Qg + (size_t)Bb * Nn * CQ;
    unsigned short* Vg = Kt + (size_t)Bb * Nn * CQ;

    SpatAttn_proj_kernel<<<(Bb * Nn) / 256, 256, 0, stream>>>(
        x, Wq, bq, Wk, bk, Wv, bv, Qg, Kt, Vg);

    SpatAttn_flash_kernel<<<Bb * (Nn / QT), 256, 0, stream>>>(
        (const unsigned int*)Qg, (const unsigned int*)Kt,
        (const unsigned int*)Vg, gamma, out);
}
